// TriGAT_91027536871852
// MI455X (gfx1250) — compile-verified, matrix operand order checked by reading, not run on hardware
//
#include <hip/hip_runtime.h>
#include <hip/hip_bf16.h>
#include <math.h>
#include <stdint.h>

typedef __attribute__((ext_vector_type(16))) _Float16 v16h;
typedef __attribute__((ext_vector_type(8)))  float    v8f;
typedef __attribute__((ext_vector_type(4)))  unsigned u32x4;
typedef __attribute__((ext_vector_type(8)))  int      i32x8;
typedef __attribute__((ext_vector_type(4)))  int      i32x4;

#define IN_DIM      512
#define HID         128
#define HEADS       4
#define OUT_DIM     64
#define NUM_CLASSES 4

#define BM 128
#define BN 64
#define BK 32

#define CDIV(a,b) (((a)+(b)-1)/(b))

#if defined(__has_builtin)
#  if __has_builtin(__builtin_amdgcn_tensor_load_to_lds) && __has_builtin(__builtin_amdgcn_s_wait_tensorcnt)
#    define HAVE_TDM 1
#  endif
#endif

// ---------- monotonic float <-> uint key (for atomicMax on floats) ----------
static __device__ __forceinline__ unsigned fkey(float f) {
    unsigned b = __float_as_uint(f);
    return b ^ ((b >> 31) ? 0xFFFFFFFFu : 0x80000000u);
}
static __device__ __forceinline__ float funkey(unsigned k) {
    unsigned b = k ^ ((k >> 31) ? 0x80000000u : 0xFFFFFFFFu);
    return __uint_as_float(b);
}
#define NEG_MAX_KEY 0x00800000u   // fkey(-FLT_MAX)

// ---------- conversion kernels ----------
__global__ void k_cvt_rows_f16(const float* __restrict__ src, _Float16* __restrict__ dst,
                               int rows_valid, int rows_pad, int cols) {
    long long i = (long long)blockIdx.x * blockDim.x + threadIdx.x;
    long long total = (long long)rows_pad * cols;
    if (i >= total) return;
    int r = (int)(i / cols);
    float v = (r < rows_valid) ? src[i] : 0.0f;
    dst[i] = (_Float16)v;
}

__global__ void k_cvt_flat_f16(const float* __restrict__ src, _Float16* __restrict__ dst, long long n) {
    long long i = (long long)blockIdx.x * blockDim.x + threadIdx.x;
    if (i < n) dst[i] = (_Float16)src[i];
}

// ---------- WMMA GEMM: C[M,Nc] = A[Mpad,K](f16) * B[K,Nc](f16), f32 out ----------
// block = 256 threads (8 waves). Block tile BM x BN, wave computes 16 x 64.
// Requires: Mpad % 128 == 0 (A zero-padded), Nc % 64 == 0, K % 32 == 0.
__global__ __launch_bounds__(256)
void k_wmma_gemm(const _Float16* __restrict__ A, const _Float16* __restrict__ B,
                 float* __restrict__ C, int M_valid, int Nc, int K) {
    __shared__ _Float16 sA[BM * BK];       // row-major [BM][BK]   (8 KB)
    __shared__ _Float16 sBt[BN * BK];      // transposed [BN][BK]  (4 KB)

    const int tid  = threadIdx.x;
    const int wave = tid >> 5;
    const int lane = tid & 31;
    const int mrow = lane & 15;
    const int hi   = (lane >> 4) & 1;

    const int rowBase = blockIdx.y * BM;
    const int colBase = blockIdx.x * BN;

    v8f acc[4] = {};

#ifdef HAVE_TDM
    // Per-wave TDM: each wave DMAs its own 16 x BK A-slice into LDS.
    const unsigned ldsA  = (unsigned)(uintptr_t)&sA[0] + (unsigned)(wave * 16 * BK * 2);
    const unsigned long long gArow =
        (unsigned long long)(uintptr_t)A +
        2ull * ((unsigned long long)(rowBase + wave * 16) * (unsigned)K);
#endif

    for (int k0 = 0; k0 < K; k0 += BK) {
#ifdef HAVE_TDM
        {   // issue tensor DMA for this wave's A slice (EXEC-independent, TENSORcnt-tracked)
            unsigned long long ga = gArow + 2ull * (unsigned)k0;
            u32x4 g0 = { 1u,                                    // count=1, user mode
                         ldsA,                                  // lds_addr (bytes)
                         (unsigned)ga,                          // global_addr[31:0]
                         (unsigned)((ga >> 32) & 0x01FFFFFFull) | (2u << 30) }; // addr[56:32] | type=2
            i32x8 g1;
            g1[0] = 0x00010000;                                 // data_size=1 (2 bytes), no flags
            g1[1] = (int)((K & 0xFFFF) << 16);                  // tensor_dim0[15:0]
            g1[2] = (int)((unsigned)K >> 16);                   // tensor_dim0[31:16] | tensor_dim1[15:0]=0
            g1[3] = (int)(0x4000u | (32u << 16));               // tensor_dim1=2^30 | tile_dim0=32
            g1[4] = 16;                                         // tile_dim1=16, tile_dim2=0
            g1[5] = (int)K;                                     // tensor_dim0_stride[31:0]
            g1[6] = 0;
            g1[7] = 0;
            i32x4 z4 = {0, 0, 0, 0};
#if __clang_major__ >= 23
            i32x8 z8 = {0, 0, 0, 0, 0, 0, 0, 0};
            __builtin_amdgcn_tensor_load_to_lds(g0, g1, z4, z4, z8, 0);
#else
            __builtin_amdgcn_tensor_load_to_lds(g0, g1, z4, z4, 0);
#endif
        }
#else
        // --- cooperative A tile load (BM x BK halves), 2 x uint4 per thread ---
#pragma unroll
        for (int it = 0; it < 2; ++it) {
            int off = tid * 8 + it * 2048;           // half index within tile
            int r = off >> 5;                        // / BK
            int c = off & (BK - 1);
            uint4 v = *(const uint4*)(A + (long long)(rowBase + r) * K + k0 + c);
            *(uint4*)&sA[off] = v;
        }
#endif
        // --- load B tile (BK x BN halves) and transpose into sBt[BN][BK] ---
        {
            int off = tid * 8;
            int kr = off >> 6;                       // / BN
            int c  = off & (BN - 1);
            uint4 v = *(const uint4*)(B + (long long)(k0 + kr) * Nc + colBase + c);
            if (k0 + BK < K)                          // prefetch next B tile
                __builtin_prefetch(B + (long long)(k0 + BK + kr) * Nc + colBase + c, 0, 1);
            _Float16 h8[8];
            *(uint4*)h8 = v;
#pragma unroll
            for (int j = 0; j < 8; ++j)
                sBt[(c + j) * BK + kr] = h8[j];
        }
#ifdef HAVE_TDM
        __builtin_amdgcn_s_wait_tensorcnt(0);        // this wave's A slice landed in LDS
#endif
        __syncthreads();

        // --- A fragment (16x32 f16, ISA layout) ---
        v16h afrag;
        {
            const _Float16* ar = &sA[(wave * 16 + mrow) * BK];
#pragma unroll
            for (int v = 0; v < 8; ++v) {
                int kp = ((v >> 2) << 3) + (hi << 2) + (v & 3);  // dword index = K/2
                unsigned w = *(const unsigned*)(ar + 2 * kp);
                afrag[2 * v]     = __builtin_bit_cast(_Float16, (unsigned short)(w & 0xFFFFu));
                afrag[2 * v + 1] = __builtin_bit_cast(_Float16, (unsigned short)(w >> 16));
            }
        }
        // --- 4 B fragments + WMMA ---
#pragma unroll
        for (int b = 0; b < 4; ++b) {
            v16h bfrag;
            const _Float16* br = &sBt[(b * 16 + mrow) * BK];
#pragma unroll
            for (int v = 0; v < 8; ++v) {
                int kp = ((v >> 2) << 3) + (hi << 2) + (v & 3);
                unsigned w = *(const unsigned*)(br + 2 * kp);
                bfrag[2 * v]     = __builtin_bit_cast(_Float16, (unsigned short)(w & 0xFFFFu));
                bfrag[2 * v + 1] = __builtin_bit_cast(_Float16, (unsigned short)(w >> 16));
            }
            acc[b] = __builtin_amdgcn_wmma_f32_16x16x32_f16(
                false, afrag, false, bfrag, (short)0, acc[b], false, false);
        }
        __syncthreads();
    }

    // --- store C: VGPR v -> M = v + 8*hi, N = lane&15 ---
    const int waveTop = rowBase + wave * 16;
    if (waveTop + 15 < M_valid) {
        // fast path: whole wave tile in range, unguarded coalesced stores
#pragma unroll
        for (int b = 0; b < 4; ++b) {
#pragma unroll
            for (int v = 0; v < 8; ++v) {
                int gr = waveTop + v + 8 * hi;
                int gc = colBase + b * 16 + mrow;
                C[(long long)gr * Nc + gc] = acc[b][v];
            }
        }
    } else {
#pragma unroll
        for (int b = 0; b < 4; ++b) {
#pragma unroll
            for (int v = 0; v < 8; ++v) {
                int gr = waveTop + v + 8 * hi;
                int gc = colBase + b * 16 + mrow;
                if (gr < M_valid) C[(long long)gr * Nc + gc] = acc[b][v];
            }
        }
    }
}

// ---------- GAT node scores: a_s[n,h] = <h[n,h,:], att_src[h,:]> ----------
__global__ void k_node_scores(const float* __restrict__ hlin,
                              const float* __restrict__ att_src,
                              const float* __restrict__ att_dst,
                              float* __restrict__ as_, float* __restrict__ ad_,
                              int N, int H, int F) {
    int i = blockIdx.x * blockDim.x + threadIdx.x;
    if (i >= N * H) return;
    int h = i % H, n = i / H;
    const float* row = hlin + ((long long)n * H + h) * F;
    const float* sv  = att_src + h * F;
    const float* dv  = att_dst + h * F;
    float a = 0.f, b = 0.f;
    for (int f = 0; f < F; ++f) { float v = row[f]; a += v * sv[f]; b += v * dv[f]; }
    as_[i] = a; ad_[i] = b;
}

// ---------- init kernels ----------
__global__ void k_init_keys(unsigned* __restrict__ p, long long n, unsigned val) {
    long long i = (long long)blockIdx.x * blockDim.x + threadIdx.x;
    if (i < n) p[i] = val;
}
__global__ void k_init_red(unsigned* red) { red[0] = NEG_MAX_KEY; ((float*)red)[1] = 0.f; }

// ---------- edge pass 1: segment max ----------
__global__ void k_edge_max(const int* __restrict__ ei, int E, int N,
                           const float* __restrict__ as_, const float* __restrict__ ad_,
                           unsigned* __restrict__ mkey, int H) {
    long long i = (long long)blockIdx.x * blockDim.x + threadIdx.x;
    long long tot = (long long)(E + N) * H;
    if (i >= tot) return;
    int h = (int)(i % H); int eid = (int)(i / H);
    int s, d;
    if (eid < E) { s = ei[eid]; d = ei[E + eid]; } else { s = d = eid - E; }
    float sc = as_[(long long)s * H + h] + ad_[(long long)d * H + h];
    sc = sc > 0.f ? sc : 0.2f * sc;
    atomicMax(&mkey[(long long)d * H + h], fkey(sc));
}

// ---------- edge pass 2: sum of exp ----------
__global__ void k_edge_sum(const int* __restrict__ ei, int E, int N,
                           const float* __restrict__ as_, const float* __restrict__ ad_,
                           const unsigned* __restrict__ mkey, float* __restrict__ ssum, int H) {
    long long i = (long long)blockIdx.x * blockDim.x + threadIdx.x;
    long long tot = (long long)(E + N) * H;
    if (i >= tot) return;
    int h = (int)(i % H); int eid = (int)(i / H);
    int s, d;
    if (eid < E) { s = ei[eid]; d = ei[E + eid]; } else { s = d = eid - E; }
    float sc = as_[(long long)s * H + h] + ad_[(long long)d * H + h];
    sc = sc > 0.f ? sc : 0.2f * sc;
    float ex = __expf(sc - funkey(mkey[(long long)d * H + h]));
    atomicAdd(&ssum[(long long)d * H + h], ex);
}

// ---------- edge pass 3: scatter messages (vec4) ----------
__global__ void k_edge_scatter(const int* __restrict__ ei, int E, int N,
                               const float* __restrict__ as_, const float* __restrict__ ad_,
                               const unsigned* __restrict__ mkey, const float* __restrict__ ssum,
                               const float* __restrict__ hsrc, float* __restrict__ accum,
                               int H, int F) {
    long long i = (long long)blockIdx.x * blockDim.x + threadIdx.x;
    int chunks = F >> 2;
    long long tot = (long long)(E + N) * H * chunks;
    if (i >= tot) return;
    int c = (int)(i % chunks);
    long long t = i / chunks;
    int h = (int)(t % H);
    int eid = (int)(t / H);
    int s, d;
    if (eid < E) { s = ei[eid]; d = ei[E + eid]; } else { s = d = eid - E; }
    long long sh = (long long)s * H + h;
    long long dh = (long long)d * H + h;
    float sc = as_[sh] + ad_[dh];
    sc = sc > 0.f ? sc : 0.2f * sc;
    float alpha = __expf(sc - funkey(mkey[dh])) / (ssum[dh] + 1e-16f);
    const float4 hv = *(const float4*)(hsrc + sh * F + (c << 2));
    float* dst = accum + dh * F + (c << 2);
    atomicAdd(dst + 0, hv.x * alpha);
    atomicAdd(dst + 1, hv.y * alpha);
    atomicAdd(dst + 2, hv.z * alpha);
    atomicAdd(dst + 3, hv.w * alpha);
}

// ---------- bias (+ optional ELU), in place ----------
__global__ void k_bias_act(float* __restrict__ x, const float* __restrict__ b,
                           long long n, int C, int do_elu) {
    long long i = (long long)blockIdx.x * blockDim.x + threadIdx.x;
    if (i >= n) return;
    int c = (int)(i % C);
    float v = x[i] + b[c];
    if (do_elu) v = v > 0.f ? v : (__expf(v) - 1.f);
    x[i] = v;
}

// ---------- scores = tanh(g3 + ba) @ wv + bv ----------
__global__ void k_tanh_score(const float* __restrict__ g3, const float* __restrict__ ba,
                             const float* __restrict__ wv, const float* __restrict__ bv,
                             float* __restrict__ scores, int N) {
    int n = blockIdx.x * blockDim.x + threadIdx.x;
    if (n >= N) return;
    const float* r = g3 + (long long)n * OUT_DIM;
    float s = bv[0];
#pragma unroll 8
    for (int f = 0; f < OUT_DIM; ++f) s += tanhf(r[f] + ba[f]) * wv[f];
    scores[n] = s;
}

// ---------- global softmax reductions over N ----------
__global__ void k_rmax(const float* __restrict__ scores, unsigned* __restrict__ red, int N) {
    int n = blockIdx.x * blockDim.x + threadIdx.x;
    if (n < N) atomicMax(&red[0], fkey(scores[n]));
}
__global__ void k_rsum(const float* __restrict__ scores, unsigned* __restrict__ red, int N) {
    int n = blockIdx.x * blockDim.x + threadIdx.x;
    if (n >= N) return;
    float gmax = funkey(red[0]);
    atomicAdd(&((float*)red)[1], __expf(scores[n] - gmax));
}

// ---------- normalize class_attn rows ----------
__global__ void k_cnorm(const float* __restrict__ ca, float* __restrict__ cn) {
    int c = threadIdx.x;
    if (c >= NUM_CLASSES) return;
    float s = 0.f;
    for (int f = 0; f < OUT_DIM; ++f) { float v = ca[c * OUT_DIM + f]; s += v * v; }
    float inv = 1.f / fmaxf(sqrtf(s), 1e-12f);
    for (int f = 0; f < OUT_DIM; ++f) cn[c * OUT_DIM + f] = ca[c * OUT_DIM + f] * inv;
}

// ---------- finalize: xw, inter, out ----------
__global__ void k_final(const float* __restrict__ h2, const float* __restrict__ scores,
                        const unsigned* __restrict__ red, const float* __restrict__ cn,
                        const float* __restrict__ Wc, const float* __restrict__ bc,
                        float* __restrict__ out, int N) {
    int n = blockIdx.x * blockDim.x + threadIdx.x;
    if (n >= N) return;
    float gmax = funkey(red[0]);
    float ssum = ((const float*)red)[1];
    float attw = __expf(scores[n] - gmax) / ssum;

    float xw[OUT_DIM];
    float nrm = 0.f;
    const float* hr = h2 + (long long)n * OUT_DIM;
#pragma unroll
    for (int f = 0; f < OUT_DIM; ++f) { float v = hr[f] * attw; xw[f] = v; nrm += v * v; }
    float inv = 1.f / fmaxf(sqrtf(nrm), 1e-12f);

    float* oxw = out + (long long)N * NUM_CLASSES + (long long)n * OUT_DIM;
#pragma unroll
    for (int f = 0; f < OUT_DIM; ++f) oxw[f] = xw[f];

    float* oin = out + (long long)N * (NUM_CLASSES + OUT_DIM) + (long long)n * NUM_CLASSES;
#pragma unroll
    for (int c = 0; c < NUM_CLASSES; ++c) {
        float s = 0.f;
#pragma unroll
        for (int f = 0; f < OUT_DIM; ++f) s += xw[f] * inv * cn[c * OUT_DIM + f];
        oin[c] = s;
    }
    float* oo = out + (long long)n * NUM_CLASSES;
#pragma unroll
    for (int c = 0; c < NUM_CLASSES; ++c) {
        float s = bc[c];
#pragma unroll
        for (int f = 0; f < OUT_DIM; ++f) s += xw[f] * Wc[f * NUM_CLASSES + c];
        oo[c] = s;
    }
}

// =======================================================================
extern "C" void kernel_launch(void* const* d_in, const int* in_sizes, int n_in,
                              void* d_out, int out_size, void* d_ws, size_t ws_size,
                              hipStream_t stream) {
    const float* x        = (const float*)d_in[0];
    const int*   ei       = (const int*)d_in[1];
    const float* W1       = (const float*)d_in[2];
    const float* att_src1 = (const float*)d_in[3];
    const float* att_dst1 = (const float*)d_in[4];
    const float* b1       = (const float*)d_in[5];
    const float* W2       = (const float*)d_in[6];
    const float* att_src2 = (const float*)d_in[7];
    const float* att_dst2 = (const float*)d_in[8];
    const float* b2       = (const float*)d_in[9];
    const float* Wa       = (const float*)d_in[10];
    const float* ba       = (const float*)d_in[11];
    const float* wv       = (const float*)d_in[12];
    const float* bv       = (const float*)d_in[13];
    const float* ca       = (const float*)d_in[14];
    const float* Wc       = (const float*)d_in[15];
    const float* bc       = (const float*)d_in[16];

    const int N    = in_sizes[0] / IN_DIM;
    const int E    = in_sizes[1] / 2;
    const int Npad = CDIV(N, BM) * BM;
    const long long Etot = (long long)E + N;

    // ---- workspace layout (256B aligned slices) ----
    char* w = (char*)d_ws;
    size_t off = 0;
    auto take = [&](size_t bytes) -> char* {
        char* p = w + off;
        off = (off + bytes + 255) & ~(size_t)255;
        return p;
    };
    _Float16* h16A = (_Float16*)take((size_t)Npad * IN_DIM * sizeof(_Float16));
    _Float16* w16  = (_Float16*)take((size_t)IN_DIM * IN_DIM * sizeof(_Float16));
    float*    lin  = (float*)take((size_t)N * IN_DIM * sizeof(float));
    float*    acc  = (float*)take((size_t)N * IN_DIM * sizeof(float));
    float*    as1  = (float*)take((size_t)N * HEADS * sizeof(float));
    float*    ad1  = (float*)take((size_t)N * HEADS * sizeof(float));
    unsigned* m1   = (unsigned*)take((size_t)N * HEADS * sizeof(unsigned));
    float*    s1   = (float*)take((size_t)N * HEADS * sizeof(float));
    float*    as2  = (float*)take((size_t)N * sizeof(float));
    float*    ad2  = (float*)take((size_t)N * sizeof(float));
    unsigned* m2   = (unsigned*)take((size_t)N * sizeof(unsigned));
    float*    s2   = (float*)take((size_t)N * sizeof(float));
    float*    sco  = (float*)take((size_t)N * sizeof(float));
    unsigned* red  = (unsigned*)take(256);
    float*    cn   = (float*)take(NUM_CLASSES * OUT_DIM * sizeof(float));

    const int T = 256;

    // ===== layer 1 =====
    {
        long long tot = (long long)Npad * IN_DIM;
        k_cvt_rows_f16<<<CDIV(tot, T), T, 0, stream>>>(x, h16A, N, Npad, IN_DIM);
        long long wn = (long long)IN_DIM * IN_DIM;
        k_cvt_flat_f16<<<CDIV(wn, T), T, 0, stream>>>(W1, w16, wn);
        dim3 g1(IN_DIM / BN, Npad / BM);
        k_wmma_gemm<<<g1, T, 0, stream>>>(h16A, w16, lin, N, IN_DIM, IN_DIM);

        k_node_scores<<<CDIV(N * HEADS, T), T, 0, stream>>>(lin, att_src1, att_dst1,
                                                            as1, ad1, N, HEADS, HID);
        k_init_keys<<<CDIV((long long)N * HEADS, T), T, 0, stream>>>(m1, (long long)N * HEADS, NEG_MAX_KEY);
        hipMemsetAsync(s1, 0, (size_t)N * HEADS * sizeof(float), stream);
        hipMemsetAsync(acc, 0, (size_t)N * IN_DIM * sizeof(float), stream);

        long long ew = Etot * HEADS;
        k_edge_max<<<CDIV(ew, T), T, 0, stream>>>(ei, E, N, as1, ad1, m1, HEADS);
        k_edge_sum<<<CDIV(ew, T), T, 0, stream>>>(ei, E, N, as1, ad1, m1, s1, HEADS);
        long long sw = Etot * HEADS * (HID >> 2);
        k_edge_scatter<<<CDIV(sw, T), T, 0, stream>>>(ei, E, N, as1, ad1, m1, s1,
                                                      lin, acc, HEADS, HID);
        k_bias_act<<<CDIV((long long)N * IN_DIM, T), T, 0, stream>>>(acc, b1,
                                                                     (long long)N * IN_DIM, IN_DIM, 1);
    }

    // ===== layer 2 =====
    {
        long long tot = (long long)Npad * IN_DIM;
        k_cvt_rows_f16<<<CDIV(tot, T), T, 0, stream>>>(acc, h16A, N, Npad, IN_DIM);
        long long wn = (long long)IN_DIM * OUT_DIM;
        k_cvt_flat_f16<<<CDIV(wn, T), T, 0, stream>>>(W2, w16, wn);
        dim3 g2(OUT_DIM / BN, Npad / BM);
        k_wmma_gemm<<<g2, T, 0, stream>>>(h16A, w16, lin, N, OUT_DIM, IN_DIM);

        k_node_scores<<<CDIV(N, T), T, 0, stream>>>(lin, att_src2, att_dst2,
                                                    as2, ad2, N, 1, OUT_DIM);
        k_init_keys<<<CDIV((long long)N, T), T, 0, stream>>>(m2, (long long)N, NEG_MAX_KEY);
        hipMemsetAsync(s2, 0, (size_t)N * sizeof(float), stream);
        hipMemsetAsync(acc, 0, (size_t)N * OUT_DIM * sizeof(float), stream);

        k_edge_max<<<CDIV(Etot, T), T, 0, stream>>>(ei, E, N, as2, ad2, m2, 1);
        k_edge_sum<<<CDIV(Etot, T), T, 0, stream>>>(ei, E, N, as2, ad2, m2, s2, 1);
        long long sw = Etot * (OUT_DIM >> 2);
        k_edge_scatter<<<CDIV(sw, T), T, 0, stream>>>(ei, E, N, as2, ad2, m2, s2,
                                                      lin, acc, 1, OUT_DIM);
        k_bias_act<<<CDIV((long long)N * OUT_DIM, T), T, 0, stream>>>(acc, b2,
                                                                      (long long)N * OUT_DIM, OUT_DIM, 0);
    }

    // ===== attention head: tanh(h2@Wa + ba)@wv + bv, softmax over N =====
    {
        long long tot = (long long)Npad * OUT_DIM;
        k_cvt_rows_f16<<<CDIV(tot, T), T, 0, stream>>>(acc, h16A, N, Npad, OUT_DIM);
        long long wn = (long long)OUT_DIM * OUT_DIM;
        k_cvt_flat_f16<<<CDIV(wn, T), T, 0, stream>>>(Wa, w16, wn);
        dim3 g3(OUT_DIM / BN, Npad / BM);
        k_wmma_gemm<<<g3, T, 0, stream>>>(h16A, w16, lin, N, OUT_DIM, OUT_DIM);

        k_tanh_score<<<CDIV(N, T), T, 0, stream>>>(lin, ba, wv, bv, sco, N);
        k_init_red<<<1, 1, 0, stream>>>(red);
        k_rmax<<<CDIV(N, T), T, 0, stream>>>(sco, red, N);
        k_rsum<<<CDIV(N, T), T, 0, stream>>>(sco, red, N);
    }

    // ===== outputs =====
    k_cnorm<<<1, 32, 0, stream>>>(ca, cn);
    k_final<<<CDIV(N, T), T, 0, stream>>>(acc, sco, red, cn, Wc, bc, (float*)d_out, N);
}
